// DescrptDPA3V1_35287451304686
// MI455X (gfx1250) — matrix operand order repaired and checked by reading, not compiled, not run on hardware
//
#include <hip/hip_runtime.h>
#include <hip/hip_bf16.h>

typedef __attribute__((ext_vector_type(16))) _Float16 v16h;
typedef __attribute__((ext_vector_type(8)))  float    v8f;

#define TPB 256

constexpr int   kNLoc   = 4096;
constexpr int   kNAll   = 6144;
constexpr int   kNEdge  = 262144;
constexpr int   kNAngle = 409600;
constexpr float kInvDynESel  = 10.0f / 64.0f;            // 1/dyn_e_sel
constexpr float kSymFactor   = (10.0f / 64.0f) / 3.0f;   // dyn_e_sel^-1 / 3
constexpr float kRsqrtDynA   = 1.0f;                     // (10/10)^-0.5

__device__ __forceinline__ float silu_f(float x) { return x / (1.0f + __expf(-x)); }

// ---- WMMA fragment loaders (wave32, f16 16x16x32) -------------------------
// A (16x32, row-major in LDS, row stride 34 halves = conflict-free):
//   lanes 0-15  : row = lane,    K in {0..15}
//   lanes 16-31 : row = lane-16, K in {16..31}
__device__ __forceinline__ v16h a_frag(const _Float16* sA, int rowBase, int lane) {
  union { v16h v; unsigned u[8]; } r;
  const int hi = (lane >> 4) & 1;
  const unsigned* base =
      (const unsigned*)(sA + (rowBase + (lane & 15)) * 34);
  #pragma unroll
  for (int j = 0; j < 4; ++j) {
    r.u[j]     = base[hi * 8 + j];          // K = hi*16 + 2j , +1
    r.u[4 + j] = base[hi * 8 + 4 + j];      // K = hi*16 + 8 + 2j , +1
  }
  return r.v;
}

// B (32x16): Wt stored as f16 [N][K] (K contiguous); lane n holds column
// nBase+(lane&15), K range kBase+{0..15} (lanes 0-15) or +{16..31} (16-31).
__device__ __forceinline__ v16h b_frag(const _Float16* Wt, int K, int nBase,
                                       int kBase, int lane) {
  const int n = nBase + (lane & 15);
  const int k = kBase + ((lane >> 4) ? 16 : 0);
  return *(const v16h*)(Wt + n * K + k);
}

// ---- weight transpose + f16 convert: dst[n*K+k] = (f16)W[k*N+n] -----------
__global__ __launch_bounds__(TPB) void wt_convert_kernel(
    const float* __restrict__ W, _Float16* __restrict__ dst, int K, int N) {
  int idx = blockIdx.x * TPB + threadIdx.x;
  if (idx >= K * N) return;
  int n = idx / K, k = idx % K;
  dst[idx] = (_Float16)W[k * N + n];
}

// ---- sym_op scatter: h2g2[i,x,c] += h2[e,x]*ebd[e,c]*sw[e] ----------------
__global__ __launch_bounds__(TPB) void sym_scatter_kernel(
    const float* __restrict__ edge_ebd, const float* __restrict__ node_ext,
    const float* __restrict__ h2, const float* __restrict__ sw,
    const int* __restrict__ n2e, const int* __restrict__ n_ext2e,
    float* __restrict__ h2g2_e, float* __restrict__ h2g2_n) {
  int idx = blockIdx.x * TPB + threadIdx.x;
  if (idx >= kNEdge * 192) return;
  int e = idx / 192, c = idx % 192;
  int i = n2e[e];
  float s = sw[e];
  float hx = h2[e * 3 + 0], hy = h2[e * 3 + 1], hz = h2[e * 3 + 2];
  float v; float* dst; int d;
  if (c < 64) { v = edge_ebd[e * 64 + c] * s;  dst = h2g2_e + i * 192 + c;        d = 64; }
  else { int cn = c - 64; v = node_ext[n_ext2e[e] * 128 + cn] * s;
         dst = h2g2_n + i * 384 + cn; d = 128; }
  atomicAdd(dst,         hx * v);
  atomicAdd(dst + d,     hy * v);
  atomicAdd(dst + 2 * d, hz * v);
}

// ---- sym_op build: g[i,a,c] = factor * sum_x h[x,a]*h[x,c] -> f16 ---------
__global__ __launch_bounds__(TPB) void sym_build_kernel(
    const float* __restrict__ h2g2_e, const float* __restrict__ h2g2_n,
    _Float16* __restrict__ sym_cat) {
  int idx = blockIdx.x * TPB + threadIdx.x;
  if (idx >= kNLoc * 768) return;
  int i = idx / 768, j = idx % 768;
  const float* h; int a, c, d;
  if (j < 256) { a = j >> 6; c = j & 63;  h = h2g2_e + i * 192; d = 64; }
  else { int j2 = j - 256; a = j2 >> 7; c = j2 & 127; h = h2g2_n + i * 384; d = 128; }
  float g = 0.f;
  #pragma unroll
  for (int x = 0; x < 3; ++x) g += h[x * d + a] * h[x * d + c];
  sym_cat[idx] = (_Float16)(g * kSymFactor);
}

// ---- edge GEMM: [node_i | nei_node | edge_ebd] (K=320) @ Wt(192x320) ------
__global__ __launch_bounds__(TPB) void edge_gemm_kernel(
    const float* __restrict__ node_ext, const float* __restrict__ edge_ebd,
    const float* __restrict__ sw, const int* __restrict__ n2e,
    const int* __restrict__ n_ext2e, const _Float16* __restrict__ Wt,
    const float* __restrict__ b_ne, const float* __restrict__ b_es,
    const float* __restrict__ e_res0, float* __restrict__ node_msg,
    float* __restrict__ e_out) {
  constexpr int K = 320, NT = 12;
  __shared__ _Float16 sA[128 * 34];
  __shared__ int sI0[128], sI1[128];
  const int tid = threadIdx.x, wave = tid >> 5, lane = tid & 31;
  const int e0 = blockIdx.x * 128;
  for (int r = tid; r < 128; r += TPB) { sI0[r] = n2e[e0 + r]; sI1[r] = n_ext2e[e0 + r]; }
  v8f acc[NT] = {};
  for (int kc = 0; kc < K / 32; ++kc) {
    __syncthreads();
    for (int li = tid; li < 128 * 32; li += TPB) {
      int row = li >> 5, kk = li & 31, k = kc * 32 + kk;
      float v;
      if (k < 128)      v = node_ext[sI0[row] * 128 + k];
      else if (k < 256) v = node_ext[sI1[row] * 128 + (k - 128)];
      else              v = edge_ebd[(e0 + row) * 64 + (k - 256)];
      sA[row * 34 + kk] = (_Float16)v;
    }
    __syncthreads();
    v16h a = a_frag(sA, wave * 16, lane);
    #pragma unroll
    for (int nt = 0; nt < NT; ++nt) {
      v16h b = b_frag(Wt, K, nt * 16, kc * 32, lane);
      acc[nt] = __builtin_amdgcn_wmma_f32_16x16x32_f16(
          false, a, false, b, (short)0, acc[nt], false, false);
    }
  }
  const int nl = lane & 15, mh = (lane >> 4) ? 8 : 0;
  #pragma unroll
  for (int nt = 0; nt < NT; ++nt) {
    int n = nt * 16 + nl;
    float bias = (n < 128) ? b_ne[n] : b_es[n - 128];
    #pragma unroll
    for (int r = 0; r < 8; ++r) {
      int m = wave * 16 + mh + r;
      int e = e0 + m;
      float s = silu_f(acc[nt][r] + bias);
      if (n < 128) {
        atomicAdd(&node_msg[sI0[m] * 128 + n], s * sw[e]);
      } else {
        int c = n - 128;
        e_out[e * 64 + c] = edge_ebd[e * 64 + c] + e_res0[c] * s;
      }
    }
  }
}

// ---- angle GEMM: [angle|node|edge_ik|edge_ij] (K=288) @ Wt(96x288) --------
__global__ __launch_bounds__(TPB) void angle_gemm_kernel(
    const float* __restrict__ angle_ebd, const float* __restrict__ node_ext,
    const float* __restrict__ edge_ebd, const float* __restrict__ a_sw,
    const int* __restrict__ n2a, const int* __restrict__ eij2a,
    const int* __restrict__ eik2a, const _Float16* __restrict__ Wt,
    const float* __restrict__ b_ea1, const float* __restrict__ b_as,
    const float* __restrict__ a_res0, float* __restrict__ reduced,
    float* __restrict__ a_out) {
  constexpr int K = 288, NT = 6;
  __shared__ _Float16 sA[128 * 34];
  __shared__ int sN[128], sIK[128], sIJ[128];
  const int tid = threadIdx.x, wave = tid >> 5, lane = tid & 31;
  const int a0 = blockIdx.x * 128;
  for (int r = tid; r < 128; r += TPB) {
    sN[r] = n2a[a0 + r]; sIK[r] = eik2a[a0 + r]; sIJ[r] = eij2a[a0 + r];
  }
  v8f acc[NT] = {};
  for (int kc = 0; kc < K / 32; ++kc) {
    __syncthreads();
    for (int li = tid; li < 128 * 32; li += TPB) {
      int row = li >> 5, kk = li & 31, k = kc * 32 + kk;
      float v;
      if (k < 32)       v = angle_ebd[(a0 + row) * 32 + k];
      else if (k < 160) v = node_ext[sN[row] * 128 + (k - 32)];
      else if (k < 224) v = edge_ebd[sIK[row] * 64 + (k - 160)];
      else              v = edge_ebd[sIJ[row] * 64 + (k - 224)];
      sA[row * 34 + kk] = (_Float16)v;
    }
    __syncthreads();
    v16h a = a_frag(sA, wave * 16, lane);
    #pragma unroll
    for (int nt = 0; nt < NT; ++nt) {
      v16h b = b_frag(Wt, K, nt * 16, kc * 32, lane);
      acc[nt] = __builtin_amdgcn_wmma_f32_16x16x32_f16(
          false, a, false, b, (short)0, acc[nt], false, false);
    }
  }
  const int nl = lane & 15, mh = (lane >> 4) ? 8 : 0;
  #pragma unroll
  for (int nt = 0; nt < NT; ++nt) {
    int n = nt * 16 + nl;
    float bias = (n < 64) ? b_ea1[n] : b_as[n - 64];
    #pragma unroll
    for (int r = 0; r < 8; ++r) {
      int m = wave * 16 + mh + r;
      int a_id = a0 + m;
      float s = silu_f(acc[nt][r] + bias);
      if (n < 64) {
        atomicAdd(&reduced[sIJ[m] * 64 + n], s * a_sw[a_id]);
      } else {
        int c = n - 64;
        a_out[a_id * 32 + c] = angle_ebd[a_id * 32 + c] + a_res0[c] * s;
      }
    }
  }
}

// ---- edge-angle-msg GEMM: reduced(262144x64) @ Wt(64x64), RMW e_out -------
__global__ __launch_bounds__(TPB) void edge_msg_kernel(
    const float* __restrict__ reduced, const _Float16* __restrict__ Wt,
    const float* __restrict__ b2, const float* __restrict__ e_res1,
    float* __restrict__ e_out) {
  constexpr int K = 64, NT = 4;
  __shared__ _Float16 sA[128 * 34];
  const int tid = threadIdx.x, wave = tid >> 5, lane = tid & 31;
  const int e0 = blockIdx.x * 128;
  v8f acc[NT] = {};
  for (int kc = 0; kc < K / 32; ++kc) {
    __syncthreads();
    for (int li = tid; li < 128 * 32; li += TPB) {
      int row = li >> 5, kk = li & 31;
      sA[row * 34 + kk] =
          (_Float16)(reduced[(e0 + row) * 64 + kc * 32 + kk] * kRsqrtDynA);
    }
    __syncthreads();
    v16h a = a_frag(sA, wave * 16, lane);
    #pragma unroll
    for (int nt = 0; nt < NT; ++nt) {
      v16h b = b_frag(Wt, K, nt * 16, kc * 32, lane);
      acc[nt] = __builtin_amdgcn_wmma_f32_16x16x32_f16(
          false, a, false, b, (short)0, acc[nt], false, false);
    }
  }
  const int nl = lane & 15, mh = (lane >> 4) ? 8 : 0;
  #pragma unroll
  for (int nt = 0; nt < NT; ++nt) {
    int n = nt * 16 + nl;
    float bias = b2[n], r1 = e_res1[n];
    #pragma unroll
    for (int r = 0; r < 8; ++r) {
      int e = e0 + wave * 16 + mh + r;
      e_out[e * 64 + n] += r1 * silu_f(acc[nt][r] + bias);
    }
  }
}

// ---- node GEMMs: self (K=128) + sym (K=768), full residual epilogue -------
__global__ __launch_bounds__(TPB) void node_gemm_kernel(
    const float* __restrict__ node_ext, const _Float16* __restrict__ sym_cat,
    const _Float16* __restrict__ wt_self, const _Float16* __restrict__ wt_sym,
    const float* __restrict__ b_self, const float* __restrict__ b_sym,
    const float* __restrict__ node_msg, const float* __restrict__ n_res0,
    const float* __restrict__ n_res1, const float* __restrict__ n_res2,
    float* __restrict__ n_out) {
  __shared__ _Float16 sA[128 * 34];
  const int tid = threadIdx.x, wave = tid >> 5, lane = tid & 31;
  const int i0 = blockIdx.x * 128;
  v8f acc1[8] = {}, acc2[8] = {};
  for (int kc = 0; kc < 4; ++kc) {               // phase 1: node_self, K=128
    __syncthreads();
    for (int li = tid; li < 128 * 32; li += TPB) {
      int row = li >> 5, kk = li & 31;
      sA[row * 34 + kk] = (_Float16)node_ext[(i0 + row) * 128 + kc * 32 + kk];
    }
    __syncthreads();
    v16h a = a_frag(sA, wave * 16, lane);
    #pragma unroll
    for (int nt = 0; nt < 8; ++nt) {
      v16h b = b_frag(wt_self, 128, nt * 16, kc * 32, lane);
      acc1[nt] = __builtin_amdgcn_wmma_f32_16x16x32_f16(
          false, a, false, b, (short)0, acc1[nt], false, false);
    }
  }
  for (int kc = 0; kc < 24; ++kc) {              // phase 2: node_sym, K=768
    __syncthreads();
    for (int li = tid; li < 128 * 32; li += TPB) {
      int row = li >> 5, kk = li & 31;
      sA[row * 34 + kk] = sym_cat[(i0 + row) * 768 + kc * 32 + kk];
    }
    __syncthreads();
    v16h a = a_frag(sA, wave * 16, lane);
    #pragma unroll
    for (int nt = 0; nt < 8; ++nt) {
      v16h b = b_frag(wt_sym, 768, nt * 16, kc * 32, lane);
      acc2[nt] = __builtin_amdgcn_wmma_f32_16x16x32_f16(
          false, a, false, b, (short)0, acc2[nt], false, false);
    }
  }
  const int nl = lane & 15, mh = (lane >> 4) ? 8 : 0;
  #pragma unroll
  for (int nt = 0; nt < 8; ++nt) {
    int n = nt * 16 + nl;
    float bs = b_self[n], by = b_sym[n];
    float r0 = n_res0[n], r1 = n_res1[n], r2 = n_res2[n];
    #pragma unroll
    for (int r = 0; r < 8; ++r) {
      int i = i0 + wave * 16 + mh + r;
      n_out[i * 128 + n] = node_ext[i * 128 + n]
                         + r0 * silu_f(acc1[nt][r] + bs)
                         + r1 * silu_f(acc2[nt][r] + by)
                         + r2 * node_msg[i * 128 + n] * kInvDynESel;
    }
  }
}

extern "C" void kernel_launch(void* const* d_in, const int* in_sizes, int n_in,
                              void* d_out, int out_size, void* d_ws, size_t ws_size,
                              hipStream_t stream) {
  (void)in_sizes; (void)n_in; (void)out_size; (void)ws_size;
  const float* node_ext   = (const float*)d_in[0];
  const float* edge_ebd   = (const float*)d_in[1];
  const float* h2         = (const float*)d_in[2];
  const float* angle_ebd  = (const float*)d_in[3];
  const float* sw         = (const float*)d_in[6];
  const float* a_sw       = (const float*)d_in[9];
  const int*   edge_index  = (const int*)d_in[10];
  const int*   angle_index = (const int*)d_in[11];
  const float* W_node_self   = (const float*)d_in[12];
  const float* b_node_self   = (const float*)d_in[13];
  const float* W_node_sym    = (const float*)d_in[14];
  const float* b_node_sym    = (const float*)d_in[15];
  const float* W_node_edge   = (const float*)d_in[16];
  const float* b_node_edge   = (const float*)d_in[17];
  const float* W_edge_self   = (const float*)d_in[18];
  const float* b_edge_self   = (const float*)d_in[19];
  const float* W_edge_angle1 = (const float*)d_in[20];
  const float* b_edge_angle1 = (const float*)d_in[21];
  const float* W_edge_angle2 = (const float*)d_in[22];
  const float* b_edge_angle2 = (const float*)d_in[23];
  const float* W_angle_self  = (const float*)d_in[24];
  const float* b_angle_self  = (const float*)d_in[25];
  const float* n_res0 = (const float*)d_in[26];
  const float* n_res1 = (const float*)d_in[27];
  const float* n_res2 = (const float*)d_in[28];
  const float* e_res0 = (const float*)d_in[29];
  const float* e_res1 = (const float*)d_in[30];
  const float* a_res0 = (const float*)d_in[31];

  const int* n2e     = edge_index;
  const int* n_ext2e = edge_index + kNEdge;
  const int* n2a   = angle_index;
  const int* eij2a = angle_index + kNAngle;
  const int* eik2a = angle_index + 2 * kNAngle;

  // workspace layout
  _Float16* wt_edge  = (_Float16*)d_ws;              // 192*320
  _Float16* wt_angle = wt_edge  + 192 * 320;         // 96*288
  _Float16* wt_msg   = wt_angle + 96 * 288;          // 64*64
  _Float16* wt_self  = wt_msg   + 64 * 64;           // 128*128
  _Float16* wt_sym   = wt_self  + 128 * 128;         // 128*768
  _Float16* sym_cat  = wt_sym   + 128 * 768;         // 4096*768
  float* h2g2_e   = (float*)(sym_cat + kNLoc * 768); // 4096*3*64
  float* h2g2_n   = h2g2_e + kNLoc * 192;            // 4096*3*128
  float* node_msg = h2g2_n + kNLoc * 384;            // 4096*128
  float* reduced  = node_msg + kNLoc * 128;          // 262144*64
  const size_t acc_bytes =
      (size_t)(kNLoc * 192 + kNLoc * 384 + kNLoc * 128 + (size_t)kNEdge * 64) * 4;

  float* n_out = (float*)d_out;
  float* e_out = n_out + kNLoc * 128;
  float* a_out = e_out + (size_t)kNEdge * 64;

  hipMemsetAsync(h2g2_e, 0, acc_bytes, stream);

  auto cdiv = [](long a, long b) { return (int)((a + b - 1) / b); };
  wt_convert_kernel<<<cdiv(320 * 128, TPB), TPB, 0, stream>>>(W_node_edge,   wt_edge,             320, 128);
  wt_convert_kernel<<<cdiv(320 * 64,  TPB), TPB, 0, stream>>>(W_edge_self,   wt_edge + 128 * 320, 320, 64);
  wt_convert_kernel<<<cdiv(288 * 64,  TPB), TPB, 0, stream>>>(W_edge_angle1, wt_angle,            288, 64);
  wt_convert_kernel<<<cdiv(288 * 32,  TPB), TPB, 0, stream>>>(W_angle_self,  wt_angle + 64 * 288, 288, 32);
  wt_convert_kernel<<<cdiv(64 * 64,   TPB), TPB, 0, stream>>>(W_edge_angle2, wt_msg,              64, 64);
  wt_convert_kernel<<<cdiv(128 * 128, TPB), TPB, 0, stream>>>(W_node_self,   wt_self,             128, 128);
  wt_convert_kernel<<<cdiv(768 * 128, TPB), TPB, 0, stream>>>(W_node_sym,    wt_sym,              768, 128);

  sym_scatter_kernel<<<cdiv((long)kNEdge * 192, TPB), TPB, 0, stream>>>(
      edge_ebd, node_ext, h2, sw, n2e, n_ext2e, h2g2_e, h2g2_n);
  sym_build_kernel<<<cdiv(kNLoc * 768, TPB), TPB, 0, stream>>>(
      h2g2_e, h2g2_n, sym_cat);

  edge_gemm_kernel<<<kNEdge / 128, TPB, 0, stream>>>(
      node_ext, edge_ebd, sw, n2e, n_ext2e, wt_edge, b_node_edge, b_edge_self,
      e_res0, node_msg, e_out);
  angle_gemm_kernel<<<kNAngle / 128, TPB, 0, stream>>>(
      angle_ebd, node_ext, edge_ebd, a_sw, n2a, eij2a, eik2a, wt_angle,
      b_edge_angle1, b_angle_self, a_res0, reduced, a_out);
  edge_msg_kernel<<<kNEdge / 128, TPB, 0, stream>>>(
      reduced, wt_msg, b_edge_angle2, e_res1, e_out);
  node_gemm_kernel<<<kNLoc / 128, TPB, 0, stream>>>(
      node_ext, sym_cat, wt_self, wt_sym, b_node_self, b_node_sym, node_msg,
      n_res0, n_res1, n_res2, n_out);
}
